// DDSP_with_features_1898375544960
// MI455X (gfx1250) — matrix operand order repaired and checked by reading, not compiled, not run on hardware
//
#include <hip/hip_runtime.h>
#include <hip/hip_bf16.h>

typedef __attribute__((ext_vector_type(16))) _Float16 v16h;
typedef __attribute__((ext_vector_type(8)))  float    v8f;

union AF { v16h h; unsigned int u[8]; };

static constexpr int kSR=16000, kBLOCK=160, kNFFT=1024, kNMELS=128, kNMFCC=30,
  kHID=512, kNHARM=100, kNBANDS=65, kB=4, kT=1000, kN=160000;
static constexpr int kROWS=kB*kT;                 // 4000 (multiple of 16)
static constexpr int kKSPEC=544;                  // 513 padded to mult of 32
static constexpr int kG3=3*kHID;                  // 1536
static constexpr int kNHARMP=112, kNBANDSP=80;    // 101/65 padded to mult of 16
static constexpr int kKOUT=544;                   // 514 padded to mult of 32
static constexpr int kIMP=16128;                  // 63*256 (16000 padded)
static constexpr int kAPRON=16384;                // zero history apron for FIR
static constexpr int kPAD=256;                    // trailing pad
static constexpr int kSTRIDE=kAPRON+kN+kPAD;      // per-batch sig row stride
static constexpr float kPI=3.14159265358979323846f;
static constexpr float kTWOPI=6.28318530717958647692f;

__device__ __forceinline__ float sigf(float x){ return 1.0f/(1.0f+__expf(-x)); }

#if defined(__gfx1250__) && __has_builtin(__builtin_amdgcn_global_load_async_to_lds_b32)
#define HAVE_ASYNC_LDS 1
#endif

#ifdef HAVE_ASYNC_LDS
__device__ __forceinline__ void async_b32(const float* g, float* l)
{
  __builtin_amdgcn_global_load_async_to_lds_b32(
      (__attribute__((address_space(1))) int*)(unsigned long long)g,
      (__attribute__((address_space(3))) int*)l, 0, 0);
}
__device__ __forceinline__ void wait_async0()
{
#if __has_builtin(__builtin_amdgcn_s_wait_asynccnt)
  __builtin_amdgcn_s_wait_asynccnt(0);
#else
  asm volatile("s_wait_asynccnt 0x0" ::: "memory");
#endif
}
#endif

// ---------------------------------------------------------------------------
// Generic WMMA GEMM: C[M x N] = A[M x K](f16,row-major) * B[K x N](f16) + bias
// one 16x16 output tile per wave; 8 waves per block; K stepped by 32.
// Fragment layouts per CDNA5 ISA 7.12.2 (wave32).
// ---------------------------------------------------------------------------
__global__ void __launch_bounds__(256) k_gemm(
    const _Float16* __restrict__ A, int lda,
    const _Float16* __restrict__ Bm, int ldb,
    const float* __restrict__ bias, int nbias,
    float* __restrict__ C, int ldc, int Ntiles, int K)
{
  int tid=threadIdx.x, lane=tid&31, w=tid>>5;
  int nt = blockIdx.x*8 + w;
  if (nt >= Ntiles) return;                 // wave-uniform
  int m = lane&15, hi=(lane>>4)<<3;
  const _Float16* arow = A + (size_t)(blockIdx.y*16+m)*lda;
  int kbl = (lane&15) + ((lane>>4)<<4);
  const _Float16* bbase = Bm + (size_t)nt*16;
  v8f acc = {0.f,0.f,0.f,0.f,0.f,0.f,0.f,0.f};
  for (int k0=0;k0<K;k0+=32){
    AF a,b;
#pragma unroll
    for (int r=0;r<8;++r){
      int kk = (r<4)?(hi+2*r):(16+hi+2*(r-4));
      a.u[r] = *reinterpret_cast<const unsigned int*>(arow + k0 + kk);
    }
    const _Float16* bcol = bbase + (lane&15);
#pragma unroll
    for (int s=0;s<16;++s) b.h[s] = bcol[(size_t)(k0 + ((s<8)?(hi+s):(16+hi+s-8)))*ldb];
    __builtin_prefetch(arow + k0 + 32, 0, 1);   // -> global_prefetch_b8
    acc = __builtin_amdgcn_wmma_f32_16x16x32_f16(false, a.h, false, b.h,
                                                 (short)0, acc, false, false);
  }
  int n = nt*16 + (lane&15);
  float bz = bias ? ((n<nbias)?bias[n]:0.0f) : 0.0f;
  int m0 = blockIdx.y*16 + ((lane>>4)<<3);
#pragma unroll
  for (int r=0;r<8;++r) C[(size_t)(m0+r)*ldc + n] = acc[r] + bz;
}

// ---------------------------------------------------------------------------
// Persistent GRU scan: one workgroup (16 waves). h (16x512 f16, rows>=4 stay 0)
// lives in LDS; each step computes gh = h @ whh^T (16x1536) with WMMA
// (96 N-tiles x 16 k-steps), then the gate math + h update.
// xp = precomputed  x@wih^T + bih  laid out [b*T+t][1536].
// ---------------------------------------------------------------------------
__global__ void __launch_bounds__(512) k_gru_scan(
  const float* __restrict__ xp, const _Float16* __restrict__ whh,
  const float* __restrict__ bhh, _Float16* __restrict__ y, int ldy, int T)
{
  __shared__ _Float16 sh[16*512];   // h as f16 (A-matrix for WMMA)
  __shared__ float    shf[4*512];   // h as f32 (exact recurrence)
  __shared__ float    sgh[4*1536];  // gh for the 4 real batch rows
  int tid=threadIdx.x, lane=tid&31, w=tid>>5;
  for (int i=tid;i<16*512;i+=512) sh[i]=(_Float16)0.f;
  for (int i=tid;i<4*512;i+=512) shf[i]=0.f;
  __syncthreads();
  int m=lane&15, hi=(lane>>4)<<3, kbl=(lane&15)+((lane>>4)<<4);
  for (int t=0;t<T;++t){
#pragma unroll 1
    for (int j=0;j<6;++j){
      int nt=w*6+j;
      v8f acc={0.f,0.f,0.f,0.f,0.f,0.f,0.f,0.f};
#pragma unroll 1
      for (int k0=0;k0<512;k0+=32){
        AF a,b;
#pragma unroll
        for (int r=0;r<8;++r){
          int kk=(r<4)?(hi+2*r):(16+hi+2*(r-4));
          a.u[r] = *reinterpret_cast<const unsigned int*>(&sh[m*512 + k0 + kk]);
        }
        const _Float16* bcol = whh + nt*16 + m;
#pragma unroll
        for (int s=0;s<16;++s) b.h[s] = bcol[(size_t)(k0 + ((s<8)?(hi+s):(16+hi+s-8)))*kG3];
        acc = __builtin_amdgcn_wmma_f32_16x16x32_f16(false,a.h,false,b.h,
                                                     (short)0,acc,false,false);
      }
      if (lane<16){
        int n=nt*16+lane;
#pragma unroll
        for (int r=0;r<4;++r) sgh[r*1536+n]=acc[r];
      }
    }
    __syncthreads();
    for (int idx=tid; idx<4*512; idx+=512){
      int mb=idx>>9, jj=idx&511;
      const float* gi = xp + (size_t)(mb*T+t)*kG3;
      float ir=gi[jj], iz=gi[512+jj], inn=gi[1024+jj];
      float hr=sgh[mb*1536+jj]     +bhh[jj];
      float hz=sgh[mb*1536+512+jj] +bhh[512+jj];
      float hn=sgh[mb*1536+1024+jj]+bhh[1024+jj];
      float r=sigf(ir+hr), z=sigf(iz+hz);
      float nn=tanhf(inn + r*hn);
      float hp=shf[mb*512+jj];
      float hnew=(1.0f-z)*nn + z*hp;
      shf[mb*512+jj]=hnew;
      sh[mb*512+jj]=(_Float16)hnew;
      y[(size_t)(mb*T+t)*ldy + jj]=(_Float16)hnew;
    }
    __syncthreads();
  }
}

// ---------------------------------------------------------------------------
// STFT front-end: one block per frame; reflect-pad, hann window, radix-2
// 1024-pt FFT in LDS, power spectrum -> f16 (513 bins, padded to 544).
// ---------------------------------------------------------------------------
__global__ void __launch_bounds__(256) k_fft(const float* __restrict__ sig,
                                             _Float16* __restrict__ spec)
{
  __shared__ float re[1024], im[1024];
  int row=blockIdx.x, b=row/kT, t=row%kT, tid=threadIdx.x;
  for (int s=0;s<4;++s){
    int j=tid+s*256;
    int o=t*kBLOCK + j - kNFFT/2;
    int oi = (o<0) ? -o : ((o>=kN) ? 2*(kN-1)-o : o);   // reflect pad
    float wv = 0.5f-0.5f*__cosf(kTWOPI*(float)j/1024.0f); // hanning(1025)[:-1]
    float v = sig[(size_t)b*kN+oi]*wv;
    int rj = (int)(__brev((unsigned)j)>>22);
    re[rj]=v; im[rj]=0.f;
  }
  for (int st=0; st<10; ++st){
    __syncthreads();
    int half=1<<st, mlen=half<<1;
    for (int q=0;q<2;++q){
      int bf=tid+q*256;
      int grp=bf>>st, pos=bf&(half-1);
      int i0=grp*mlen+pos, i1=i0+half;
      float ang = -kTWOPI*(float)pos/(float)mlen;
      float swv,cwv; __sincosf(ang,&swv,&cwv);
      float xr=re[i1], xi=im[i1];
      float tr=xr*cwv-xi*swv, ti=xr*swv+xi*cwv;
      float ur=re[i0], ui=im[i0];
      re[i0]=ur+tr; im[i0]=ui+ti;
      re[i1]=ur-tr; im[i1]=ui-ti;
    }
  }
  __syncthreads();
  for (int s=0;s<3;++s){
    int f=tid+s*256;
    if (f<513) spec[(size_t)row*kKSPEC+f]=(_Float16)(re[f]*re[f]+im[f]*im[f]);
    else if (f<kKSPEC) spec[(size_t)row*kKSPEC+f]=(_Float16)0.f;
  }
}

// ---------------------------------------------------------------------------
// Constant tables: mel filterbank [K=544][N=128] and DCT^T [128][32] as f16.
// ---------------------------------------------------------------------------
__device__ __forceinline__ float mel2hz(float mm){ return 700.0f*(exp10f(mm/2595.0f)-1.0f); }

__global__ void k_init_tables(_Float16* __restrict__ melfb, _Float16* __restrict__ dct)
{
  int idx=blockIdx.x*blockDim.x+threadIdx.x;
  const float melmax = 2595.0f*log10f(1.0f+8000.0f/700.0f);
  if (idx < kKSPEC*kNMELS){
    int f=idx/kNMELS, i=idx%kNMELS;
    float v=0.f;
    if (f<513){
      float fhz = (float)f*(8000.0f/512.0f);
      float l=mel2hz(melmax*(float)(i  )/(float)(kNMELS+1));
      float c=mel2hz(melmax*(float)(i+1)/(float)(kNMELS+1));
      float r=mel2hz(melmax*(float)(i+2)/(float)(kNMELS+1));
      v=fmaxf(0.0f, fminf((fhz-l)/(c-l),(r-fhz)/(r-c)));
    }
    melfb[idx]=(_Float16)v;
  } else if (idx < kKSPEC*kNMELS + 128*32){
    int k2=idx-kKSPEC*kNMELS;
    int k=k2/32, n=k2%32;
    float v=0.f;
    if (n<kNMFCC){
      v = __cosf(kPI/128.0f*((float)k+0.5f)*(float)n)*sqrtf(2.0f/128.0f);
      if (n==0) v *= 0.70710678118f;
    }
    dct[k2]=(_Float16)v;
  }
}

// Weight transpose+convert: w[dout][din] (f32) -> B-matrix [Kp][Np] (f16, 0 pad)
__global__ void k_wT(const float* __restrict__ w, _Float16* __restrict__ Bh,
                     int din, int dout, int Kp, int Np)
{
  int idx=blockIdx.x*blockDim.x+threadIdx.x;
  if (idx>=Kp*Np) return;
  int k=idx/Np, n=idx%Np;
  Bh[idx] = (k<din && n<dout) ? (_Float16)w[(size_t)n*din+k] : (_Float16)0.f;
}

// Fuse dense_z (512->16) and in_z (16->512): Wc[k][n] = sum_z Wi[n][z]*Wd[z][k]
__global__ void k_zcomb(const float* __restrict__ dzw, const float* __restrict__ dzb,
                        const float* __restrict__ izw, const float* __restrict__ izb,
                        _Float16* __restrict__ Wc, float* __restrict__ bc)
{
  int idx=blockIdx.x*blockDim.x+threadIdx.x;
  if (idx>=512*512) return;
  int k=idx/512, n=idx%512;
  float a=0.f;
#pragma unroll
  for (int z=0;z<16;++z) a += izw[n*16+z]*dzw[z*512+k];
  Wc[(size_t)k*512+n]=(_Float16)a;
  if (k==0){
    float bb=izb[n];
#pragma unroll
    for (int z=0;z<16;++z) bb += izw[n*16+z]*dzb[z];
    bc[n]=bb;
  }
}

__global__ void k_log_h(const float* __restrict__ X, _Float16* __restrict__ Y, int n)
{
  int i=blockIdx.x*blockDim.x+threadIdx.x;
  if (i<n) Y[i]=(_Float16)__logf(X[i]+1e-6f);
}

// mfcc normalization over time per (batch, coeff); pads coeffs 30..31 to 0.
__global__ void __launch_bounds__(256) k_mfcc_norm(const float* __restrict__ mf,
                                                   _Float16* __restrict__ out)
{
  __shared__ float red[256];
  int b=blockIdx.x>>5, c=blockIdx.x&31, tid=threadIdx.x;
  if (c>=kNMFCC){
    for (int t=tid;t<kT;t+=256) out[(size_t)(b*kT+t)*32+c]=(_Float16)0.f;
    return;
  }
  float s=0.f;
  for (int t=tid;t<kT;t+=256) s += mf[(size_t)(b*kT+t)*32+c];
  red[tid]=s; __syncthreads();
  for (int o=128;o>0;o>>=1){ if(tid<o) red[tid]+=red[tid+o]; __syncthreads(); }
  float mu=red[0]/(float)kT; __syncthreads();
  float v=0.f;
  for (int t=tid;t<kT;t+=256){ float d=mf[(size_t)(b*kT+t)*32+c]-mu; v+=d*d; }
  red[tid]=v; __syncthreads();
  for (int o=128;o>0;o>>=1){ if(tid<o) red[tid]+=red[tid+o]; __syncthreads(); }
  float inv=rsqrtf(red[0]/(float)kT + 1e-5f);
  for (int t=tid;t<kT;t+=256){
    float d=mf[(size_t)(b*kT+t)*32+c]-mu;
    out[(size_t)(b*kT+t)*32+c]=(_Float16)(d*inv);
  }
}

// LayerNorm (biased var) + leaky_relu(0.01), f32 in [row][512] -> f16 out strided
__global__ void __launch_bounds__(512) k_ln_lrelu(const float* __restrict__ X,
  const float* __restrict__ g, const float* __restrict__ beta,
  _Float16* __restrict__ Y, int ldy, int coloff)
{
  __shared__ float red[512];
  int row=blockIdx.x, tid=threadIdx.x;
  float x = X[(size_t)row*512+tid];
  red[tid]=x; __syncthreads();
  for (int o=256;o>0;o>>=1){ if(tid<o) red[tid]+=red[tid+o]; __syncthreads(); }
  float mu=red[0]*(1.0f/512.0f); __syncthreads();
  float d=x-mu; red[tid]=d*d; __syncthreads();
  for (int o=256;o>0;o>>=1){ if(tid<o) red[tid]+=red[tid+o]; __syncthreads(); }
  float inv=rsqrtf(red[0]*(1.0f/512.0f)+1e-5f);
  float v=d*inv*g[tid]+beta[tid];
  v = v>0.f ? v : 0.01f*v;
  Y[(size_t)row*ldy+coloff+tid]=(_Float16)v;
}

// mlp layer0 (din=1): x = inp[row]*w[n]+b[n], then LN + lrelu
__global__ void __launch_bounds__(512) k_mlp_l0(const float* __restrict__ inp,
  const float* __restrict__ w, const float* __restrict__ b,
  const float* __restrict__ g, const float* __restrict__ beta,
  _Float16* __restrict__ Y, int ldy, int coloff)
{
  __shared__ float red[512];
  int row=blockIdx.x, tid=threadIdx.x;
  float x = inp[row]*w[tid]+b[tid];
  red[tid]=x; __syncthreads();
  for (int o=256;o>0;o>>=1){ if(tid<o) red[tid]+=red[tid+o]; __syncthreads(); }
  float mu=red[0]*(1.0f/512.0f); __syncthreads();
  float d=x-mu; red[tid]=d*d; __syncthreads();
  for (int o=256;o>0;o>>=1){ if(tid<o) red[tid]+=red[tid+o]; __syncthreads(); }
  float inv=rsqrtf(red[0]*(1.0f/512.0f)+1e-5f);
  float v=d*inv*g[tid]+beta[tid];
  v = v>0.f ? v : 0.01f*v;
  Y[(size_t)row*ldy+coloff+tid]=(_Float16)v;
}

__global__ void k_copy_f2h(const float* __restrict__ X, _Float16* __restrict__ Y,
                           int ldy, int coloff, int ncol, int nrows)
{
  int i=blockIdx.x*blockDim.x+threadIdx.x;
  if (i>=nrows*ncol) return;
  int r=i/ncol, c=i%ncol;
  Y[(size_t)r*ldy+coloff+c]=(_Float16)X[(size_t)r*ncol+c];
}

// fill out_mlp input cols 512 (pitch), 513 (loudness), 514..543 (zero)
__global__ void k_fill_out(const float* __restrict__ pitch, const float* __restrict__ loud,
                           _Float16* __restrict__ Y)
{
  int i=blockIdx.x*blockDim.x+threadIdx.x;
  if (i>=kROWS*32) return;
  int r=i/32, c=i%32;
  float v = (c==0)?pitch[r]:((c==1)?loud[r]:0.f);
  Y[(size_t)r*kKOUT+512+c]=(_Float16)v;
}

// scale_fn: 2*sigmoid(x+shift)^2.302585 + 1e-7 (zero padded cols)
__global__ void k_scale(float* __restrict__ X, int ld, int nvalid, float shift, int nrows)
{
  int i=blockIdx.x*blockDim.x+threadIdx.x;
  if (i>=nrows*ld) return;
  int c=i%ld;
  if (c<nvalid){
    float s=sigf(X[i]+shift);
    X[i]=2.0f*__powf(s,2.302585092994046f)+1e-7f;
  } else X[i]=0.f;
}

// harmonic amp prep: antialias mask, normalize, phase step per frame
__global__ void k_harm_prep(const float* __restrict__ H, const float* __restrict__ pitch,
  float* __restrict__ amps, float* __restrict__ wstep, float* __restrict__ bsum)
{
  int r=blockIdx.x*blockDim.x+threadIdx.x;
  if (r>=kROWS) return;
  float p=pitch[r];
  float tot=H[(size_t)r*kNHARMP];
  float s=0.f;
  for (int k=1;k<=kNHARM;++k){
    float aa = (p*(float)k < 8000.0f) ? 1.0001f : 0.0001f;
    float a = H[(size_t)r*kNHARMP+k]*aa;
    amps[(size_t)r*kNHARM + k-1] = a;
    s += a;
  }
  float sc = tot/s;
  for (int k=0;k<kNHARM;++k) amps[(size_t)r*kNHARM+k]*=sc;
  float wv = kTWOPI*p/(float)kSR;
  wstep[r]=wv; bsum[r]=wv*(float)kBLOCK;
}

// exclusive scan of per-frame phase increments (1000 per batch, trivial)
__global__ void k_phase_scan(const float* __restrict__ bsum, float* __restrict__ offs)
{
  int b=blockIdx.x;
  if (threadIdx.x==0){
    float acc=0.f;
    for (int t=0;t<kT;++t){ offs[b*kT+t]=acc; acc+=bsum[b*kT+t]; }
  }
}

// zero the leading apron + trailing pad of each sig row (graph-replay safe)
__global__ void k_zero_apron(float* __restrict__ sigbase)
{
  int i=blockIdx.x*blockDim.x+threadIdx.x;
  const int per = kAPRON + kPAD;
  if (i>=kB*per) return;
  int b=i/per, q=i%per;
  size_t idx = (size_t)b*kSTRIDE + ((q<kAPRON) ? q : (kAPRON + kN + (q - kAPRON)));
  sigbase[idx]=0.f;
}

// harmonic bank: sin(k*omega) via Chebyshev recurrence (2 FMA per harmonic)
__global__ void __launch_bounds__(256) k_harm_synth(const float* __restrict__ amps,
  const float* __restrict__ wstep, const float* __restrict__ offs,
  float* __restrict__ sigbase)
{
  int b=blockIdx.y;
  int n=blockIdx.x*256+threadIdx.x;
  int t=n/kBLOCK, i=n%kBLOCK;
  int r=b*kT+t;
  float om = offs[r] + (float)(i+1)*wstep[r];
  float alpha=2.0f*cosf(om);
  float s1=sinf(om), s0=0.f, acc=0.f;
  const float* ar = amps + (size_t)r*kNHARM;
#pragma unroll 4
  for (int k=0;k<kNHARM;++k){
    acc += ar[k]*s1;
    float s2 = alpha*s1 - s0;
    s0=s1; s1=s2;
  }
  sigbase[(size_t)b*kSTRIDE + kAPRON + n]=acc;
}

// per-frame filtered noise: 65-band inverse DFT -> windowed IR -> causal conv
__global__ void __launch_bounds__(160) k_noise_synth(const float* __restrict__ PN,
                                                     float* __restrict__ sigbase)
{
  __shared__ float ir1[128], impk[160], nz[160];
  int r=blockIdx.x, tid=threadIdx.x;
  int b=r/kT, t=r%kT;
  const float* pn = PN + (size_t)r*kNBANDSP;
  float irv=0.f;
  if (tid<128){
    float acc=pn[0];
    for (int k=1;k<64;++k) acc += 2.0f*pn[k]*__cosf(kTWOPI*(float)k*(float)tid/128.0f);
    acc += pn[64]*__cosf(kPI*(float)tid);
    irv=acc*(1.0f/128.0f);
  }
  if (tid<128) ir1[tid]=irv;           // ir0 staged
  __syncthreads();
  float w1=0.f;
  if (tid<128){
    float h = 0.5f-0.5f*__cosf(kTWOPI*(float)tid/127.0f);   // np.hanning(128)
    w1 = ir1[(tid+64)&127]*h;          // roll +64, window
  }
  __syncthreads();
  if (tid<128) ir1[tid]=w1;
  __syncthreads();
  { int src=(tid+64)%160; impk[tid] = (src<128)?ir1[src]:0.f; }  // pad+roll -64
  { unsigned u = (unsigned)(r*160+tid);
    u = u*747796405u + 2891336453u;
    u = ((u >> ((u>>28)+4u)) ^ u)*277803737u;
    u = (u>>22)^u;
    nz[tid] = (float)u*(2.0f/4294967296.0f)-1.0f; }
  __syncthreads();
  float acc=0.f;
  for (int m2=0;m2<=tid;++m2) acc += nz[m2]*impk[tid-m2];
  sigbase[(size_t)b*kSTRIDE + kAPRON + t*kBLOCK + tid] += acc;
}

__global__ void k_imp(const float* __restrict__ rnoise, const float* __restrict__ rdecay,
                      const float* __restrict__ rwet, float* __restrict__ imp)
{
  int m=blockIdx.x*blockDim.x+threadIdx.x;
  if (m>=kIMP) return;
  float sp = log1pf(__expf(-rdecay[0]));   // softplus(-decay)
  float sw = sigf(rwet[0]);
  float v;
  if (m==0) v=1.0f;
  else if (m<kSR) v = rnoise[m]*__expf(-sp*((float)m/(float)kSR)*500.0f)*sw;
  else v=0.0f;
  imp[m]=v;
}

// ---------------------------------------------------------------------------
// reverb: causal FIR, 16128 taps, LDS-tiled (256 out samples / block).
// With the zero apron the history window is always in-bounds, so chunks are
// staged into LDS with CDNA5 async-to-LDS copies (ASYNCcnt) when available.
// ---------------------------------------------------------------------------
__global__ void __launch_bounds__(256) k_reverb(const float* __restrict__ sigbase,
  const float* __restrict__ imp, float* __restrict__ out)
{
  __shared__ float sk[256];
  __shared__ float swd[512];
  int b=blockIdx.y, n0=blockIdx.x*256, tid=threadIdx.x;
  const float* srow = sigbase + (size_t)b*kSTRIDE + kAPRON;
  float acc=0.f;
  for (int c=0;c<kIMP/256;++c){
    int kb=c*256;
    int base=n0-kb-255;
    __syncthreads();                 // previous chunk fully consumed
#ifdef HAVE_ASYNC_LDS
    async_b32(imp+kb+tid,        &sk[tid]);
    async_b32(srow+base+tid,     &swd[tid]);
    async_b32(srow+base+256+tid, &swd[256+tid]);
    wait_async0();
#else
    sk[tid]=imp[kb+tid];
    swd[tid]     = srow[base+tid];
    swd[256+tid] = srow[base+256+tid];
#endif
    __syncthreads();
#pragma unroll 8
    for (int rr=0;rr<256;++rr) acc += sk[rr]*swd[tid+255-rr];
  }
  out[(size_t)b*kN + n0 + tid]=acc;
}

// ---------------------------------------------------------------------------
extern "C" void kernel_launch(void* const* d_in, const int* in_sizes, int n_in,
                              void* d_out, int out_size, void* d_ws, size_t ws_size,
                              hipStream_t stream)
{
  const float* signal=(const float*)d_in[0];
  const float* pitch =(const float*)d_in[1];
  const float* loud  =(const float*)d_in[2];
  const float* zg_wih=(const float*)d_in[3];
  const float* zg_whh=(const float*)d_in[4];
  const float* zg_bih=(const float*)d_in[5];
  const float* zg_bhh=(const float*)d_in[6];
  const float* dz_w=(const float*)d_in[7];
  const float* dz_b=(const float*)d_in[8];
  const float* pl_w[3]={(const float*)d_in[9],(const float*)d_in[13],(const float*)d_in[17]};
  const float* pl_b[3]={(const float*)d_in[10],(const float*)d_in[14],(const float*)d_in[18]};
  const float* pl_g[3]={(const float*)d_in[11],(const float*)d_in[15],(const float*)d_in[19]};
  const float* pl_be[3]={(const float*)d_in[12],(const float*)d_in[16],(const float*)d_in[20]};
  const float* iz_w=(const float*)d_in[21];
  const float* iz_b=(const float*)d_in[22];
  const float* dg_wih=(const float*)d_in[23];
  const float* dg_whh=(const float*)d_in[24];
  const float* dg_bih=(const float*)d_in[25];
  const float* dg_bhh=(const float*)d_in[26];
  const float* om_w[3]={(const float*)d_in[27],(const float*)d_in[31],(const float*)d_in[35]};
  const float* om_b[3]={(const float*)d_in[28],(const float*)d_in[32],(const float*)d_in[36]};
  const float* om_g[3]={(const float*)d_in[29],(const float*)d_in[33],(const float*)d_in[37]};
  const float* om_be[3]={(const float*)d_in[30],(const float*)d_in[34],(const float*)d_in[38]};
  const float* th_w=(const float*)d_in[39];
  const float* th_b=(const float*)d_in[40];
  const float* tn_w=(const float*)d_in[41];
  const float* tn_b=(const float*)d_in[42];
  const float* rnoise=(const float*)d_in[43];
  const float* rdecay=(const float*)d_in[44];
  const float* rwet  =(const float*)d_in[45];
  float* out=(float*)d_out;
  (void)in_sizes; (void)n_in; (void)out_size; (void)ws_size;

  // ---- workspace bump allocator (deterministic layout) ----
  char* p=(char*)d_ws;
  auto alloc=[&](size_t bytes)->char*{ char* q=p; p += (bytes+255)&~(size_t)255; return q; };
  _Float16* spec_h  =(_Float16*)alloc((size_t)kROWS*kKSPEC*2);
  _Float16* melfb_h =(_Float16*)alloc((size_t)kKSPEC*kNMELS*2);
  _Float16* dct_h   =(_Float16*)alloc((size_t)128*32*2);
  float*    mel_f   =(float*)   alloc((size_t)kROWS*kNMELS*4);
  _Float16* logmel_h=(_Float16*)alloc((size_t)kROWS*kNMELS*2);
  float*    mfcc_f  =(float*)   alloc((size_t)kROWS*32*4);
  _Float16* mfccn_h =(_Float16*)alloc((size_t)kROWS*32*2);
  _Float16* wihz_h  =(_Float16*)alloc((size_t)32*kG3*2);
  _Float16* whhz_h  =(_Float16*)alloc((size_t)512*kG3*2);
  _Float16* zwc_h   =(_Float16*)alloc((size_t)512*512*2);
  float*    zbc     =(float*)   alloc((size_t)512*4);
  _Float16* plw1_h  =(_Float16*)alloc((size_t)512*512*2);
  _Float16* plw2_h  =(_Float16*)alloc((size_t)512*512*2);
  _Float16* dwih_h  =(_Float16*)alloc((size_t)kG3*kG3*2);
  _Float16* dwhh_h  =(_Float16*)alloc((size_t)512*kG3*2);
  _Float16* om0_h   =(_Float16*)alloc((size_t)kKOUT*512*2);
  _Float16* om1_h   =(_Float16*)alloc((size_t)512*512*2);
  _Float16* om2_h   =(_Float16*)alloc((size_t)512*512*2);
  _Float16* thw_h   =(_Float16*)alloc((size_t)512*kNHARMP*2);
  _Float16* tnw_h   =(_Float16*)alloc((size_t)512*kNBANDSP*2);
  float*    xproj   =(float*)   alloc((size_t)kROWS*kG3*4);   // shared z/dec
  float*    tmpF    =(float*)   alloc((size_t)kROWS*512*4);
  _Float16* actA_h  =(_Float16*)alloc((size_t)kROWS*512*2);
  _Float16* actB_h  =(_Float16*)alloc((size_t)kROWS*512*2);
  _Float16* zy_h    =(_Float16*)alloc((size_t)kROWS*512*2);
  _Float16* decin_h =(_Float16*)alloc((size_t)kROWS*kG3*2);
  _Float16* outin_h =(_Float16*)alloc((size_t)kROWS*kKOUT*2);
  float*    harm_f  =(float*)   alloc((size_t)kROWS*kNHARMP*4);
  float*    noise_f =(float*)   alloc((size_t)kROWS*kNBANDSP*4);
  float*    hamps   =(float*)   alloc((size_t)kROWS*kNHARM*4);
  float*    wstep   =(float*)   alloc((size_t)kROWS*4);
  float*    bsum    =(float*)   alloc((size_t)kROWS*4);
  float*    offs    =(float*)   alloc((size_t)kROWS*4);
  float*    sigb    =(float*)   alloc((size_t)kB*kSTRIDE*4);  // apron + sig + pad
  float*    imp     =(float*)   alloc((size_t)kIMP*4);

  auto gemm=[&](const _Float16* A,int lda,const _Float16* Bm,int ldb,
                const float* bias,int nbias,float* C,int ldc,int M,int Nn,int K){
    int Nt=Nn/16;
    dim3 g((Nt+7)/8, M/16);
    k_gemm<<<g,256,0,stream>>>(A,lda,Bm,ldb,bias,nbias,C,ldc,Nt,K);
  };
  auto wT=[&](const float* w,_Float16* dst,int din,int dout,int Kp,int Np){
    int tot=Kp*Np;
    k_wT<<<(tot+255)/256,256,0,stream>>>(w,dst,din,dout,Kp,Np);
  };

  // ---- constant tables + weight conversion ----
  { int tot=kKSPEC*kNMELS+128*32;
    k_init_tables<<<(tot+255)/256,256,0,stream>>>(melfb_h,dct_h); }
  wT(zg_wih, wihz_h, kNMFCC, kG3, 32, kG3);
  wT(zg_whh, whhz_h, kHID, kG3, 512, kG3);
  wT(pl_w[1], plw1_h, 512,512,512,512);
  wT(pl_w[2], plw2_h, 512,512,512,512);
  wT(dg_wih, dwih_h, kG3, kG3, kG3, kG3);
  wT(dg_whh, dwhh_h, kHID, kG3, 512, kG3);
  wT(om_w[0], om0_h, 514,512,kKOUT,512);
  wT(om_w[1], om1_h, 512,512,512,512);
  wT(om_w[2], om2_h, 512,512,512,512);
  wT(th_w, thw_h, 512, kNHARM+1, 512, kNHARMP);
  wT(tn_w, tnw_h, 512, kNBANDS, 512, kNBANDSP);
  k_zcomb<<<(512*512+255)/256,256,0,stream>>>(dz_w,dz_b,iz_w,iz_b,zwc_h,zbc);
  k_zero_apron<<<((kB*(kAPRON+kPAD))+255)/256,256,0,stream>>>(sigb);

  // ---- feature front-end ----
  k_fft<<<kROWS,256,0,stream>>>(signal, spec_h);
  gemm(spec_h,kKSPEC, melfb_h,kNMELS, nullptr,0, mel_f,kNMELS, kROWS,kNMELS,kKSPEC);
  k_log_h<<<(kROWS*kNMELS+255)/256,256,0,stream>>>(mel_f, logmel_h, kROWS*kNMELS);
  gemm(logmel_h,kNMELS, dct_h,32, nullptr,0, mfcc_f,32, kROWS,32,128);
  k_mfcc_norm<<<kB*32,256,0,stream>>>(mfcc_f, mfccn_h);

  // ---- z GRU ----
  gemm(mfccn_h,32, wihz_h,kG3, zg_bih,kG3, xproj,kG3, kROWS,kG3,32);
  k_gru_scan<<<1,512,0,stream>>>(xproj, whhz_h, zg_bhh, zy_h, 512, kT);
  gemm(zy_h,512, zwc_h,512, zbc,512, tmpF,512, kROWS,512,512);  // fused dense_z+in_z
  k_copy_f2h<<<(kROWS*512+255)/256,256,0,stream>>>(tmpF, decin_h, kG3, 1024, 512, kROWS);

  // ---- mlp_pl (pitch then loudness), final layer writes into decoder concat ----
  k_mlp_l0<<<kROWS,512,0,stream>>>(pitch, pl_w[0], pl_b[0], pl_g[0], pl_be[0], actA_h, 512, 0);
  gemm(actA_h,512, plw1_h,512, pl_b[1],512, tmpF,512, kROWS,512,512);
  k_ln_lrelu<<<kROWS,512,0,stream>>>(tmpF, pl_g[1], pl_be[1], actB_h, 512, 0);
  gemm(actB_h,512, plw2_h,512, pl_b[2],512, tmpF,512, kROWS,512,512);
  k_ln_lrelu<<<kROWS,512,0,stream>>>(tmpF, pl_g[2], pl_be[2], decin_h, kG3, 0);

  k_mlp_l0<<<kROWS,512,0,stream>>>(loud, pl_w[0], pl_b[0], pl_g[0], pl_be[0], actA_h, 512, 0);
  gemm(actA_h,512, plw1_h,512, pl_b[1],512, tmpF,512, kROWS,512,512);
  k_ln_lrelu<<<kROWS,512,0,stream>>>(tmpF, pl_g[1], pl_be[1], actB_h, 512, 0);
  gemm(actB_h,512, plw2_h,512, pl_b[2],512, tmpF,512, kROWS,512,512);
  k_ln_lrelu<<<kROWS,512,0,stream>>>(tmpF, pl_g[2], pl_be[2], decin_h, kG3, 512);

  // ---- decoder GRU ----
  gemm(decin_h,kG3, dwih_h,kG3, dg_bih,kG3, xproj,kG3, kROWS,kG3,kG3);
  k_gru_scan<<<1,512,0,stream>>>(xproj, dwhh_h, dg_bhh, outin_h, kKOUT, kT);
  k_fill_out<<<(kROWS*32+255)/256,256,0,stream>>>(pitch, loud, outin_h);

  // ---- out_mlp ----
  gemm(outin_h,kKOUT, om0_h,512, om_b[0],512, tmpF,512, kROWS,512,kKOUT);
  k_ln_lrelu<<<kROWS,512,0,stream>>>(tmpF, om_g[0], om_be[0], actA_h, 512, 0);
  gemm(actA_h,512, om1_h,512, om_b[1],512, tmpF,512, kROWS,512,512);
  k_ln_lrelu<<<kROWS,512,0,stream>>>(tmpF, om_g[1], om_be[1], actB_h, 512, 0);
  gemm(actB_h,512, om2_h,512, om_b[2],512, tmpF,512, kROWS,512,512);
  k_ln_lrelu<<<kROWS,512,0,stream>>>(tmpF, om_g[2], om_be[2], actA_h, 512, 0);

  // ---- heads ----
  gemm(actA_h,512, thw_h,kNHARMP, th_b,kNHARM+1, harm_f,kNHARMP, kROWS,kNHARMP,512);
  k_scale<<<(kROWS*kNHARMP+255)/256,256,0,stream>>>(harm_f,kNHARMP,kNHARM+1,0.0f,kROWS);
  gemm(actA_h,512, tnw_h,kNBANDSP, tn_b,kNBANDS, noise_f,kNBANDSP, kROWS,kNBANDSP,512);
  k_scale<<<(kROWS*kNBANDSP+255)/256,256,0,stream>>>(noise_f,kNBANDSP,kNBANDS,-5.0f,kROWS);

  // ---- synthesis ----
  k_harm_prep<<<(kROWS+127)/128,128,0,stream>>>(harm_f, pitch, hamps, wstep, bsum);
  k_phase_scan<<<kB,32,0,stream>>>(bsum, offs);
  { dim3 g(kN/256, kB); k_harm_synth<<<g,256,0,stream>>>(hamps, wstep, offs, sigb); }
  k_noise_synth<<<kROWS,160,0,stream>>>(noise_f, sigb);

  // ---- reverb ----
  k_imp<<<kIMP/256,256,0,stream>>>(rnoise, rdecay, rwet, imp);
  { dim3 g(kN/256, kB); k_reverb<<<g,256,0,stream>>>(sigb, imp, out); }
}
